// SelfAttention_25512105739133
// MI455X (gfx1250) — matrix operand order repaired and checked by reading, not compiled, not run on hardware
//
#include <hip/hip_runtime.h>
#include <hip/hip_bf16.h>

typedef __attribute__((ext_vector_type(16))) _Float16 v16h;
typedef __attribute__((ext_vector_type(8)))  _Float16 v8h;
typedef __attribute__((ext_vector_type(8)))  float    v8f;

#define BATCH 4
#define NPIX  4096
#define CDIM  256
#define CKDIM 32

// ---------------- Kernel 0: weight convert + transpose to f16 ----------------
// Wf/Wg [C,CK] f32 -> WfT/WgT [CK,C] f16 ; Wh [C,C] f32 -> WhT [C,C] f16 (transposed)
// Transposed layout makes every WMMA B-fragment a contiguous 32B per-lane load.
__global__ void wcvt_kernel(const float* __restrict__ Wf, const float* __restrict__ Wg,
                            const float* __restrict__ Wh,
                            _Float16* __restrict__ WfT, _Float16* __restrict__ WgT,
                            _Float16* __restrict__ WhT) {
  const int i = blockIdx.x * blockDim.x + threadIdx.x;   // 0 .. C*C-1
  const int o = i >> 8;                                  // out channel
  const int c = i & (CDIM - 1);                          // in channel
  WhT[i] = (_Float16)Wh[(size_t)c * CDIM + o];
  if (i < CKDIM * CDIM) {
    WfT[i] = (_Float16)Wf[(size_t)c * CKDIM + o];
    WgT[i] = (_Float16)Wg[(size_t)c * CKDIM + o];
  }
}

// ---------------- Kernel 1: WMMA projections ----------------
// One block = 16 pixels, 2 waves.
//  each wave: 8 h-channel-tiles (64 wmma) ; wave0 also f (16 wmma), wave1 g.
// h is produced directly in transposed f16 layout hT[B,C,N] (lane = channel,
// VGPR = pixel -> one contiguous 16B store per lane per tile).
// f/g go through a 2KB LDS transpose to row-major f16 [B*N, 32].
__global__ void __launch_bounds__(64)
proj_wmma(const float* __restrict__ x,
          const _Float16* __restrict__ WfT, const _Float16* __restrict__ WgT,
          const _Float16* __restrict__ WhT,
          const float* __restrict__ bf, const float* __restrict__ bg,
          const float* __restrict__ bh,
          _Float16* __restrict__ fK, _Float16* __restrict__ gQ,
          _Float16* __restrict__ hT) {
  __shared__ __align__(64) _Float16 lds[2][16][CKDIM];

  const int lane  = threadIdx.x & 31;
  const int wave  = threadIdx.x >> 5;
  const int pix0  = blockIdx.x * 16;        // global pixel row-block
  const int batch = blockIdx.x >> 8;
  const int pl    = (blockIdx.x & 255) * 16; // pixel offset within batch
  const int n     = lane & 15;
  const int khalf = (lane & 16) ? 16 : 0;
  const int qsel  = (lane & 16) ? 8 : 0;
  const int cbase = wave * 128;

  // ---- build 8 A-fragments of x (16 pixels x 256 ch), f32 -> f16 ----
  const float* xrow = x + (size_t)(pix0 + n) * CDIM + ((lane & 16) ? 16 : 0);
  v16h xa[8];
#pragma unroll
  for (int kf = 0; kf < 8; ++kf) {
    v8f lo = *(const v8f*)(xrow + 32 * kf);
    v8f hi = *(const v8f*)(xrow + 32 * kf + 8);
#pragma unroll
    for (int i = 0; i < 8; ++i) {
      xa[kf][i]     = (_Float16)lo[i];
      xa[kf][8 + i] = (_Float16)hi[i];
    }
  }

  // ---- h = x @ Wh + bh : 8 tiles per wave ----
#pragma unroll
  for (int t = 0; t < 8; ++t) {
    const int c = cbase + t * 16 + n;
    v8f hacc;
#pragma unroll
    for (int r = 0; r < 8; ++r) hacc[r] = 0.0f;
#pragma unroll
    for (int kf = 0; kf < 8; ++kf) {
      v16h wb = *(const v16h*)(WhT + (size_t)c * CDIM + 32 * kf + khalf);
      hacc = __builtin_amdgcn_wmma_f32_16x16x32_f16(false, xa[kf], false, wb,
                                                    (short)0, hacc, false, false);
    }
    const float bc = bh[c];
    v8h hh;
#pragma unroll
    for (int r = 0; r < 8; ++r) hh[r] = (_Float16)(hacc[r] + bc);
    // transposed store: contiguous 8 pixels for this channel
    *(v8h*)(hT + ((size_t)batch * CDIM + c) * NPIX + pl + qsel) = hh;
  }

  // ---- f (wave0) / g (wave1) = x @ W + b : 2 tiles of 16 ch ----
  const _Float16* WpT = (wave == 0) ? WfT : WgT;
  const float*    bp  = (wave == 0) ? bf  : bg;
#pragma unroll
  for (int t = 0; t < 2; ++t) {
    v8f facc;
#pragma unroll
    for (int r = 0; r < 8; ++r) facc[r] = 0.0f;
#pragma unroll
    for (int kf = 0; kf < 8; ++kf) {
      v16h wb = *(const v16h*)(WpT + (size_t)(t * 16 + n) * CDIM + 32 * kf + khalf);
      facc = __builtin_amdgcn_wmma_f32_16x16x32_f16(false, xa[kf], false, wb,
                                                    (short)0, facc, false, false);
    }
    const float bc = bp[t * 16 + n];
#pragma unroll
    for (int r = 0; r < 8; ++r)
      lds[wave][qsel + r][t * 16 + n] = (_Float16)(facc[r] + bc);
  }
  __syncthreads();

  // LDS -> row-major global: each lane stores half a 32-ch row (32B aligned)
  {
    const int row = lane >> 1;
    const int off = (lane & 1) * 16;
    _Float16* dst = ((wave == 0) ? fK : gQ) + (size_t)(pix0 + row) * CKDIM + off;
    *(v16h*)dst = *(const v16h*)(&lds[wave][row][off]);
  }
}

// A-fragment load for 16x32 f16 matrix stored row-major with row stride CKDIM.
__device__ inline v16h load_a_frag(const _Float16* __restrict__ tile, int lane) {
  const int m  = lane & 15;
  const int c0 = (lane & 16) ? 16 : 0;
  const _Float16* p = tile + m * CKDIM + c0;
  v8h lo = *(const v8h*)(p);
  v8h hi = *(const v8h*)(p + 8);
  v16h a;
#pragma unroll
  for (int i = 0; i < 8; ++i) { a[i] = lo[i]; a[8 + i] = hi[i]; }
  return a;
}

// ---------------- Kernel 2: fused flash attention + residual ----------------
__global__ void __launch_bounds__(64)
attn_kernel(const float* __restrict__ x, const float* __restrict__ gamma_p,
            const _Float16* __restrict__ fK, const _Float16* __restrict__ gQ,
            const _Float16* __restrict__ hT, float* __restrict__ out) {
  const int lane  = threadIdx.x & 31;
  const int wave  = threadIdx.x >> 5;
  const int blk   = blockIdx.x;             // B * (N/16) blocks
  const int batch = blk >> 8;
  const int q0    = (blk & 255) * 16;
  const int cbase = wave * 128;

  const int qn    = lane & 15;
  const int khalf = (lane & 16) ? 16 : 0;
  const int qsel  = (lane & 16) ? 8 : 0;

  v16h gB = *(const v16h*)(gQ + ((size_t)(batch * NPIX + q0 + qn)) * CKDIM + khalf);

  v8f acc[8];
#pragma unroll
  for (int t = 0; t < 8; ++t)
#pragma unroll
    for (int r = 0; r < 8; ++r) acc[t][r] = 0.0f;

  v8f zeroC;
#pragma unroll
  for (int r = 0; r < 8; ++r) zeroC[r] = 0.0f;

  float mrun = -__builtin_inff();
  float lrun = 0.0f;

  for (int j = 0; j < NPIX; j += 32) {
    const _Float16* fbase = fK + ((size_t)(batch * NPIX + j)) * CKDIM;
    v16h fA0 = load_a_frag(fbase, lane);
    v16h fA1 = load_a_frag(fbase + 16 * CKDIM, lane);
    if (j + 32 < NPIX)  // warm cache for next key block's f tiles
      __builtin_prefetch(fbase + 32 * CKDIM + (lane & 15) * CKDIM, 0, 3);

    v8f T0 = __builtin_amdgcn_wmma_f32_16x16x32_f16(false, fA0, false, gB,
                                                    (short)0, zeroC, false, false);
    v8f T1 = __builtin_amdgcn_wmma_f32_16x16x32_f16(false, fA1, false, gB,
                                                    (short)0, zeroC, false, false);

    float tmax = -__builtin_inff();
#pragma unroll
    for (int r = 0; r < 8; ++r) {
      tmax = fmaxf(tmax, T0[r]);
      tmax = fmaxf(tmax, T1[r]);
    }
    tmax = fmaxf(tmax, __shfl_xor(tmax, 16, 32));
    const float mnew    = fmaxf(mrun, tmax);
    const float rescale = __expf(mrun - mnew);
    mrun = mnew;

    float psum = 0.0f;
    v16h pA;
#pragma unroll
    for (int r = 0; r < 8; ++r) {
      float p0 = __expf(T0[r] - mnew);
      float p1 = __expf(T1[r] - mnew);
      psum += p0 + p1;
      pA[r]     = (_Float16)p0;
      pA[8 + r] = (_Float16)p1;
    }
    psum += __shfl_xor(psum, 16, 32);
    lrun = lrun * rescale + psum;

    float sc[8];
#pragma unroll
    for (int r = 0; r < 8; ++r) sc[r] = __shfl(rescale, qsel + r, 32);

    const _Float16* hbase =
        hT + ((size_t)batch * CDIM + cbase + qn) * NPIX + j + qsel;
#pragma unroll
    for (int t = 0; t < 8; ++t) {
#pragma unroll
      for (int r = 0; r < 8; ++r) acc[t][r] *= sc[r];
      const _Float16* hp = hbase + (size_t)t * 16 * NPIX;
      v8h hlo = *(const v8h*)(hp);
      v8h hhi = *(const v8h*)(hp + 16);
      v16h hB;
#pragma unroll
      for (int i = 0; i < 8; ++i) { hB[i] = hlo[i]; hB[8 + i] = hhi[i]; }
      acc[t] = __builtin_amdgcn_wmma_f32_16x16x32_f16(false, pA, false, hB,
                                                      (short)0, acc[t], false, false);
    }
  }

  float inv[8];
#pragma unroll
  for (int r = 0; r < 8; ++r) inv[r] = 1.0f / __shfl(lrun, qsel + r, 32);
  const float gam = gamma_p[0];
#pragma unroll
  for (int t = 0; t < 8; ++t) {
    const int c = cbase + t * 16 + qn;
#pragma unroll
    for (int r = 0; r < 8; ++r) {
      const int m = qsel + r;
      const size_t idx = ((size_t)(batch * NPIX + q0 + m)) * CDIM + c;
      out[idx] = fmaf(gam * inv[r], acc[t][r], x[idx]);
    }
  }
}

extern "C" void kernel_launch(void* const* d_in, const int* in_sizes, int n_in,
                              void* d_out, int out_size, void* d_ws, size_t ws_size,
                              hipStream_t stream) {
  (void)in_sizes; (void)n_in; (void)out_size; (void)ws_size;
  const float* x     = (const float*)d_in[0];
  const float* Wf    = (const float*)d_in[1];
  const float* bf    = (const float*)d_in[2];
  const float* Wg    = (const float*)d_in[3];
  const float* bg    = (const float*)d_in[4];
  const float* Wh    = (const float*)d_in[5];
  const float* bh    = (const float*)d_in[6];
  const float* gamma = (const float*)d_in[7];
  float* out = (float*)d_out;

  // workspace: f16 f|g [B*N,32], hT [B,C,N], WfT|WgT [CK,C], WhT [C,C]  (~10.7 MB)
  _Float16* fK  = (_Float16*)d_ws;
  _Float16* gQ  = fK  + (size_t)BATCH * NPIX * CKDIM;
  _Float16* hT  = gQ  + (size_t)BATCH * NPIX * CKDIM;
  _Float16* WfT = hT  + (size_t)BATCH * CDIM * NPIX;
  _Float16* WgT = WfT + (size_t)CKDIM * CDIM;
  _Float16* WhT = WgT + (size_t)CKDIM * CDIM;

  wcvt_kernel<<<(CDIM * CDIM) / 256, 256, 0, stream>>>(Wf, Wg, Wh, WfT, WgT, WhT);
  proj_wmma<<<BATCH * (NPIX / 16), 64, 0, stream>>>(x, WfT, WgT, WhT, bf, bg, bh,
                                                    fK, gQ, hT);
  attn_kernel<<<BATCH * (NPIX / 16), 64, 0, stream>>>(x, gamma, fK, gQ, hT, out);
}
